// FlashSkyworkLayer_78331613545193
// MI455X (gfx1250) — compile-verified, matrix operand order checked by reading, not run on hardware
//
#include <hip/hip_runtime.h>
#include <stdint.h>

// ---------------- problem constants (match reference) ----------------
#define Bsz   2
#define Sq    2048
#define Ttok  (Bsz*Sq)          // 4096
#define NH    16
#define NKV   4
#define HD    128
#define Dm    2048
#define NE    8
#define Fm    1408

typedef __bf16 bf16;
typedef __attribute__((ext_vector_type(16))) __bf16 v16bf;
typedef __attribute__((ext_vector_type(8)))  __bf16 v8bf;
typedef __attribute__((ext_vector_type(8)))  float  v8f;

#define AS1 __attribute__((address_space(1)))
#define AS3 __attribute__((address_space(3)))

// ---- CDNA5 data-mover feature gates (probe-safe) ----
#if defined(__has_builtin)
# if __has_builtin(__builtin_amdgcn_global_load_async_to_lds_b128) && \
     __has_builtin(__builtin_amdgcn_s_wait_asynccnt)
#  define USE_ASYNC_LDS 1
# endif
# if __has_builtin(__builtin_amdgcn_tensor_load_to_lds) && \
     __has_builtin(__builtin_amdgcn_s_wait_tensorcnt)
#  define USE_TDM 1
# endif
#endif

#if defined(USE_ASYNC_LDS)
// builtin signature (per hipcc diagnostic): (v4i32 AS1*, v4i32 AS3*, imm, cpol)
typedef int v4i_gv __attribute__((vector_size(16)));
// one 16-byte async global->LDS copy (tracked by ASYNCcnt)
__device__ __forceinline__ void async_cp16(const void* g, void* l) {
  __builtin_amdgcn_global_load_async_to_lds_b128((AS1 v4i_gv*)g, (AS3 v4i_gv*)l, 0, 0);
}
__device__ __forceinline__ void async_wait0() {
  __builtin_amdgcn_s_wait_asynccnt(0);
}
#endif

#if defined(USE_TDM)
typedef unsigned int u32x4 __attribute__((ext_vector_type(4)));
typedef int          i32x4v __attribute__((ext_vector_type(4)));
typedef int          i32x8v __attribute__((ext_vector_type(8)));

// Issue a TDM 2D tile load: tile0 elements (contiguous, 2B each) x tile1 rows,
// row pitch stride0 elements, packed contiguously into LDS at lds_off.
// D# layout per CDNA5 ISA 8.3/8.4 (count=1, type=2 "image", data_size=2B).
__device__ __forceinline__ void tdm_load_tile_2d(const void* gsrc, unsigned lds_off,
                                                 unsigned tdim0, unsigned tdim1,
                                                 unsigned tile0, unsigned tile1,
                                                 unsigned stride0) {
  unsigned long long ga = (unsigned long long)(uintptr_t)gsrc;
  u32x4 g0;
  g0[0] = 1u;                                          // count=1, user D#
  g0[1] = lds_off;                                     // lds_addr (bytes)
  g0[2] = (unsigned)(ga & 0xFFFFFFFFu);                // global_addr[31:0]
  g0[3] = (unsigned)((ga >> 32) & 0x1FFFFFFu)          // global_addr[56:32]
          | (2u << 30);                                // type=2
  i32x8v g1;
  g1[0] = 0x00010000;                                  // data_size=1 (2 bytes)
  g1[1] = (int)((tdim0 & 0xFFFFu) << 16);              // tensor_dim0[15:0]
  g1[2] = (int)(((tdim0 >> 16) & 0xFFFFu) |            // tensor_dim0[31:16]
                ((tdim1 & 0xFFFFu) << 16));            // tensor_dim1[15:0]
  g1[3] = (int)(((tdim1 >> 16) & 0xFFFFu) |            // tensor_dim1[31:16]
                ((tile0 & 0xFFFFu) << 16));            // tile_dim0
  g1[4] = (int)(tile1 & 0xFFFFu);                      // tile_dim1 (tile_dim2=0)
  g1[5] = (int)stride0;                                // tensor_dim0_stride[31:0]
  g1[6] = 0;
  g1[7] = 0;
  i32x4v z4 = {0, 0, 0, 0};
#if __clang_major__ >= 23
  i32x8v z8 = {0, 0, 0, 0, 0, 0, 0, 0};
  __builtin_amdgcn_tensor_load_to_lds(g0, g1, z4, z4, z8, 0);
#else
  __builtin_amdgcn_tensor_load_to_lds(g0, g1, z4, z4, 0);
#endif
}
__device__ __forceinline__ void tdm_wait0() {
  __builtin_amdgcn_s_wait_tensorcnt(0);
}
#endif

union FragU { v16bf v; v8bf h[2]; };

// Load a 16-bit WMMA operand fragment from a row-major [row][ldk] tile.
// A lane holds K = {koff..koff+7, koff+16..koff+23} (koff 0 / 8 by half-wave):
// two contiguous 16-byte loads.
__device__ __forceinline__ v16bf frag_ld(const bf16* base, int row, int ldk, int koff) {
  FragU u;
  u.h[0] = *(const v8bf*)(base + (long)row * ldk + koff);
  u.h[1] = *(const v8bf*)(base + (long)row * ldk + koff + 16);
  return u.v;
}

__device__ __forceinline__ v8f wmma_bf16(v16bf a, v16bf b, v8f c) {
  return __builtin_amdgcn_wmma_f32_16x16x32_bf16(false, a, false, b, (short)0, c,
                                                 false, false);
}

__device__ __forceinline__ v8f vzero8() {
  v8f z = {0.f, 0.f, 0.f, 0.f, 0.f, 0.f, 0.f, 0.f};
  return z;
}

// ---------------- fp32 -> bf16 weight conversion ----------------
__global__ void cvt_kernel(const float* __restrict__ src, bf16* __restrict__ dst, long n) {
  long i = (long)blockIdx.x * blockDim.x + threadIdx.x;
  long stride = (long)gridDim.x * blockDim.x;
  for (; i < n; i += stride) dst[i] = (bf16)src[i];
}

// ---------------- fused add-residual + RMSNorm ----------------
__global__ void __launch_bounds__(256)
rmsnorm_res_kernel(const float* __restrict__ x, const float* __restrict__ resin,
                   const float* __restrict__ w, float* __restrict__ resout,
                   bf16* __restrict__ normed) {
  __shared__ float red[256];
  int tok = blockIdx.x;
  const float* xr = x + (long)tok * Dm;
  const float* rr = resin + (long)tok * Dm;
  float* ro = resout + (long)tok * Dm;
  bf16* no = normed + (long)tok * Dm;
  float rv[8];
  float ss = 0.f;
#pragma unroll
  for (int i = 0; i < 8; ++i) {
    int d = threadIdx.x + i * 256;
    float r = xr[d] + rr[d];
    rv[i] = r;
    ro[d] = r;
    ss += r * r;
  }
  red[threadIdx.x] = ss;
  __syncthreads();
  for (int s = 128; s > 0; s >>= 1) {
    if (threadIdx.x < (unsigned)s) red[threadIdx.x] += red[threadIdx.x + s];
    __syncthreads();
  }
  float inv = rsqrtf(red[0] / (float)Dm + 1e-6f);
#pragma unroll
  for (int i = 0; i < 8; ++i) {
    int d = threadIdx.x + i * 256;
    no[d] = (bf16)(rv[i] * inv * w[d]);
  }
}

// ---------------- generic bf16 WMMA GEMM ----------------
// C[M,N] = A[M,K] @ B[K,N]; block tile 128x64, 8 waves of 32x32.
// A tile is fetched by the Tensor Data Mover when available.
#define BM 128
#define BN 64
#define KST 32

__global__ void __launch_bounds__(256)
gemm_bf16_kernel(const bf16* __restrict__ Ag, const bf16* __restrict__ Bg,
                 float* __restrict__ Cf, bf16* __restrict__ Cb,
                 int M, int N, int K) {
  __shared__ bf16 ldsA[BM * KST];   // [m][k]
  __shared__ bf16 ldsB[BN * KST];   // transposed: [n][k]
  int t = threadIdx.x;
  int lane = t & 31, wid = t >> 5;
  int wm = (wid & 3) * 32;
  int wn = (wid >> 2) * 32;
  long mbase = (long)blockIdx.y * BM;
  long nbase = (long)blockIdx.x * BN;
  int koff = (lane >> 4) * 8;
  int mrow = lane & 15;

#if defined(USE_TDM)
  unsigned ldsA_off = (unsigned)(uintptr_t)(AS3 bf16*)ldsA;
#endif

  v8f acc[2][2];
#pragma unroll
  for (int i = 0; i < 2; ++i)
#pragma unroll
    for (int j = 0; j < 2; ++j) acc[i][j] = vzero8();

  for (int k0 = 0; k0 < K; k0 += KST) {
    // ---- A tile 128x32 ----
#if defined(USE_TDM)
    if (wid == 0) {
      // TDM: 32-elem rows (64B) x 128 rows, pitch K, packed into ldsA
      tdm_load_tile_2d(&Ag[mbase * (long)K + k0], ldsA_off,
                       (unsigned)K, (unsigned)M, KST, BM, (unsigned)K);
    }
#elif defined(USE_ASYNC_LDS)
#pragma unroll
    for (int p = 0; p < 2; ++p) {
      int idx = t + p * 256;
      int r = idx >> 2;
      int c = (idx & 3) * 8;
      async_cp16(&Ag[(mbase + r) * (long)K + k0 + c], &ldsA[r * KST + c]);
    }
#else
#pragma unroll
    for (int p = 0; p < 2; ++p) {
      int idx = t + p * 256;
      int r = idx >> 2;
      int c = (idx & 3) * 8;
      *(v8bf*)&ldsA[r * KST + c] =
          *(const v8bf*)&Ag[(mbase + r) * (long)K + k0 + c];
    }
#endif
    // ---- B tile 32x64, transposed into [n][k] ----
    {
      int kk = t >> 3;                // 0..31
      int n0 = (t & 7) * 8;           // 0..56
      v8bf bv = *(const v8bf*)&Bg[(long)(k0 + kk) * N + nbase + n0];
#pragma unroll
      for (int i = 0; i < 8; ++i) ldsB[(n0 + i) * KST + kk] = bv[i];
    }
#if defined(USE_TDM)
    if (wid == 0) tdm_wait0();
#elif defined(USE_ASYNC_LDS)
    async_wait0();
#endif
    __syncthreads();
    if (k0 + KST < K)
      __builtin_prefetch(&Ag[(mbase + (t >> 1)) * (long)K + k0 + KST], 0, 1);

    v16bf af[2], bfr[2];
    af[0] = frag_ld(ldsA, wm + mrow, KST, koff);
    af[1] = frag_ld(ldsA, wm + 16 + mrow, KST, koff);
    bfr[0] = frag_ld(ldsB, wn + mrow, KST, koff);
    bfr[1] = frag_ld(ldsB, wn + 16 + mrow, KST, koff);
#pragma unroll
    for (int i = 0; i < 2; ++i)
#pragma unroll
      for (int j = 0; j < 2; ++j) acc[i][j] = wmma_bf16(af[i], bfr[j], acc[i][j]);
    __syncthreads();
  }

  int rbase = (lane >> 4) * 8;
  int ncol = lane & 15;
#pragma unroll
  for (int i = 0; i < 2; ++i)
#pragma unroll
    for (int j = 0; j < 2; ++j)
#pragma unroll
      for (int r = 0; r < 8; ++r) {
        long m = mbase + wm + i * 16 + rbase + r;
        long n = nbase + wn + j * 16 + ncol;
        if (Cf) Cf[m * N + n] = acc[i][j][r];
        else    Cb[m * N + n] = (bf16)acc[i][j][r];
      }
}

// ---------------- RoPE (in-place on bf16 q or k) ----------------
__global__ void rope_kernel(bf16* __restrict__ x, const float* __restrict__ cos_t,
                            const float* __restrict__ sin_t, int heads, long total) {
  long i = (long)blockIdx.x * blockDim.x + threadIdx.x;
  if (i >= total) return;
  int d = (int)(i % 64);
  long rem = i / 64;
  int hh = (int)(rem % heads);
  long tok = rem / heads;
  bf16* p = x + (tok * heads + hh) * HD;
  float c = cos_t[tok * 64 + d];
  float s = sin_t[tok * 64 + d];
  float x1 = (float)p[d];
  float x2 = (float)p[d + 64];
  p[d]      = (bf16)(x1 * c - x2 * s);
  p[d + 64] = (bf16)(x2 * c + x1 * s);
}

// ---------------- causal flash attention (GQA) ----------------
// grid (B*NH, Sq/64); 4 waves, each owns 16 query rows.
__global__ void __launch_bounds__(128)
flash_attn_kernel(const bf16* __restrict__ q, const bf16* __restrict__ k,
                  const bf16* __restrict__ v, bf16* __restrict__ o) {
  __shared__ bf16 ldsK[32 * HD];       // [kv][hd]
  __shared__ bf16 ldsV[HD * 32];       // transposed: [hd][kv]
  __shared__ bf16 ldsP[4 * 16 * 32];   // per-wave P staging
  int t = threadIdx.x, lane = t & 31, wid = t >> 5;
  int bh = blockIdx.x;
  int b = bh / NH, h = bh % NH;
  int kvh = h / (NH / NKV);
  int qtile = blockIdx.y;
  int qrow0 = qtile * 64 + wid * 16;
  int mrow = lane & 15;
  int koff = (lane >> 4) * 8;
  int rbase = (lane >> 4) * 8;
  const float scale = 0.08838834764831845f;  // 1/sqrt(128)

  long tq = (long)b * Sq + qrow0 + mrow;
  const bf16* qrow = q + tq * (NH * HD) + h * HD;
  v16bf qf[4];
#pragma unroll
  for (int kk = 0; kk < 4; ++kk) qf[kk] = frag_ld(qrow, 0, 0, kk * 32 + koff);

  v8f oacc[8];
#pragma unroll
  for (int jt = 0; jt < 8; ++jt) oacc[jt] = vzero8();
  float mi[8], li[8];
#pragma unroll
  for (int r = 0; r < 8; ++r) { mi[r] = -1e30f; li[r] = 0.f; }

  int kend = qtile * 64 + 64;
  for (int kb = 0; kb < kend; kb += 32) {
    // cooperative K/V tile load; K rows go async to LDS, V is transposed
#pragma unroll
    for (int p = 0; p < 4; ++p) {
      int idx = t + p * 128;            // 0..511
      int kr = idx >> 4;                // 0..31
      int c = (idx & 15) * 8;           // 0..120
      long tk = (long)b * Sq + kb + kr;
#if defined(USE_ASYNC_LDS)
      async_cp16(&k[tk * (NKV * HD) + kvh * HD + c], &ldsK[kr * HD + c]);
#else
      v8bf kv8 = *(const v8bf*)&k[tk * (NKV * HD) + kvh * HD + c];
      *(v8bf*)&ldsK[kr * HD + c] = kv8;
#endif
      v8bf vv8 = *(const v8bf*)&v[tk * (NKV * HD) + kvh * HD + c];
#pragma unroll
      for (int i = 0; i < 8; ++i) ldsV[(c + i) * 32 + kr] = vv8[i];
    }
#if defined(USE_ASYNC_LDS)
    async_wait0();
#endif
    __syncthreads();

    // S = Q @ K^T  (two 16-wide kv fragments)
    v8f sa[2];
    sa[0] = vzero8(); sa[1] = vzero8();
#pragma unroll
    for (int kk = 0; kk < 4; ++kk) {
      v16bf bk0 = frag_ld(ldsK, mrow,      HD, kk * 32 + koff);
      v16bf bk1 = frag_ld(ldsK, 16 + mrow, HD, kk * 32 + koff);
      sa[0] = wmma_bf16(qf[kk], bk0, sa[0]);
      sa[1] = wmma_bf16(qf[kk], bk1, sa[1]);
    }

    // causal mask + online softmax
    float pr[2][8];
#pragma unroll
    for (int r = 0; r < 8; ++r) {
      int qg = qrow0 + rbase + r;
      float v0 = sa[0][r] * scale;
      float v1 = sa[1][r] * scale;
      if (kb + mrow > qg)      v0 = -1e30f;
      if (kb + 16 + mrow > qg) v1 = -1e30f;
      float mx = fmaxf(v0, v1);
#pragma unroll
      for (int sh = 8; sh >= 1; sh >>= 1) mx = fmaxf(mx, __shfl_xor(mx, sh, 32));
      float mn = fmaxf(mi[r], mx);
      float alpha = __expf(mi[r] - mn);
      float p0 = __expf(v0 - mn);
      float p1 = __expf(v1 - mn);
      float rs = p0 + p1;
#pragma unroll
      for (int sh = 8; sh >= 1; sh >>= 1) rs += __shfl_xor(rs, sh, 32);
      li[r] = li[r] * alpha + rs;
      mi[r] = mn;
      pr[0][r] = p0; pr[1][r] = p1;
#pragma unroll
      for (int jt = 0; jt < 8; ++jt) oacc[jt][r] *= alpha;
    }

    // P: C-layout -> A-layout via LDS
    bf16* Pw = &ldsP[wid * 16 * 32];
#pragma unroll
    for (int j = 0; j < 2; ++j)
#pragma unroll
      for (int r = 0; r < 8; ++r)
        Pw[(rbase + r) * 32 + j * 16 + mrow] = (bf16)pr[j][r];
    __syncthreads();

    // O += P @ V
    v16bf pf = frag_ld(Pw, mrow, 32, koff);
#pragma unroll
    for (int jt = 0; jt < 8; ++jt) {
      v16bf bv = frag_ld(ldsV, jt * 16 + mrow, 32, koff);
      oacc[jt] = wmma_bf16(pf, bv, oacc[jt]);
    }
    __syncthreads();
  }

#pragma unroll
  for (int r = 0; r < 8; ++r) {
    float invl = 1.0f / li[r];
    long tqr = (long)b * Sq + qrow0 + rbase + r;
#pragma unroll
    for (int jt = 0; jt < 8; ++jt)
      o[tqr * (NH * HD) + h * HD + jt * 16 + mrow] = (bf16)(oacc[jt][r] * invl);
  }
}

// ---------------- router: softmax top-2, build per-expert token lists -----
__global__ void __launch_bounds__(256)
router_kernel(const bf16* __restrict__ x, const float* __restrict__ rw,
              int* __restrict__ cnts, int* __restrict__ toks,
              float* __restrict__ wts) {
  __shared__ float red[256 * NE];
  int tok = blockIdx.x;
  float acc[NE];
#pragma unroll
  for (int e = 0; e < NE; ++e) acc[e] = 0.f;
  for (int d = threadIdx.x; d < Dm; d += 256) {
    float xv = (float)x[(long)tok * Dm + d];
#pragma unroll
    for (int e = 0; e < NE; ++e) acc[e] += xv * rw[d * NE + e];
  }
#pragma unroll
  for (int e = 0; e < NE; ++e) red[threadIdx.x * NE + e] = acc[e];
  __syncthreads();
  for (int s = 128; s > 0; s >>= 1) {
    if (threadIdx.x < (unsigned)s)
#pragma unroll
      for (int e = 0; e < NE; ++e)
        red[threadIdx.x * NE + e] += red[(threadIdx.x + s) * NE + e];
    __syncthreads();
  }
  if (threadIdx.x == 0) {
    float lg[NE];
    float mx = -1e30f;
#pragma unroll
    for (int e = 0; e < NE; ++e) { lg[e] = red[e]; mx = fmaxf(mx, lg[e]); }
#pragma unroll
    for (int e = 0; e < NE; ++e) lg[e] = __expf(lg[e] - mx);
    int i0 = 0;
#pragma unroll
    for (int e = 1; e < NE; ++e) if (lg[e] > lg[i0]) i0 = e;
    int i1 = (i0 == 0) ? 1 : 0;
#pragma unroll
    for (int e = 0; e < NE; ++e) if (e != i0 && lg[e] > lg[i1]) i1 = e;
    float s2 = lg[i0] + lg[i1];
    float w0 = lg[i0] / s2, w1 = lg[i1] / s2;
    int s0 = atomicAdd(&cnts[i0], 1);
    toks[i0 * Ttok + s0] = tok; wts[i0 * Ttok + s0] = w0;
    int s1 = atomicAdd(&cnts[i1], 1);
    toks[i1 * Ttok + s1] = tok; wts[i1 * Ttok + s1] = w1;
  }
}

// ---------------- MoE grouped gate/up GEMM with fused SiLU*up ----------
__global__ void __launch_bounds__(256)
moe_gateup_kernel(const bf16* __restrict__ X, const bf16* __restrict__ W1,
                  const bf16* __restrict__ W3, const int* __restrict__ cnts,
                  const int* __restrict__ toks, bf16* __restrict__ H) {
  int e = blockIdx.z;
  int cnt = cnts[e];
  long mbase = (long)blockIdx.y * BM;
  if (mbase >= cnt) return;
  __shared__ bf16 ldsA[BM * KST];
  __shared__ bf16 ldsB1[BN * KST];
  __shared__ bf16 ldsB2[BN * KST];
  const int* etok = toks + (long)e * Ttok;
  const bf16* B1 = W1 + (long)e * Dm * Fm;
  const bf16* B2 = W3 + (long)e * Dm * Fm;
  int t = threadIdx.x;
  int lane = t & 31, wid = t >> 5;
  int wm = (wid & 3) * 32, wn = (wid >> 2) * 32;
  long nbase = (long)blockIdx.x * BN;
  int koff = (lane >> 4) * 8;
  int mrow = lane & 15;

  v8f accg[2][2], accu[2][2];
#pragma unroll
  for (int i = 0; i < 2; ++i)
#pragma unroll
    for (int j = 0; j < 2; ++j) { accg[i][j] = vzero8(); accu[i][j] = vzero8(); }

  for (int k0 = 0; k0 < Dm; k0 += KST) {
    // gathered A tile (per-lane addresses -> async LDS copies when available)
#pragma unroll
    for (int p = 0; p < 2; ++p) {
      int idx = t + p * 256;
      int r = idx >> 2;
      int c = (idx & 3) * 8;
      int tokr = (mbase + r < cnt) ? etok[mbase + r] : 0;
#if defined(USE_ASYNC_LDS)
      async_cp16(&X[(long)tokr * Dm + k0 + c], &ldsA[r * KST + c]);
#else
      *(v8bf*)&ldsA[r * KST + c] = *(const v8bf*)&X[(long)tokr * Dm + k0 + c];
#endif
    }
    {
      int kk = t >> 3;
      int n0 = (t & 7) * 8;
      v8bf b1 = *(const v8bf*)&B1[(long)(k0 + kk) * Fm + nbase + n0];
      v8bf b2 = *(const v8bf*)&B2[(long)(k0 + kk) * Fm + nbase + n0];
#pragma unroll
      for (int i = 0; i < 8; ++i) {
        ldsB1[(n0 + i) * KST + kk] = b1[i];
        ldsB2[(n0 + i) * KST + kk] = b2[i];
      }
    }
#if defined(USE_ASYNC_LDS)
    async_wait0();
#endif
    __syncthreads();
    v16bf af[2], b1f[2], b2f[2];
    af[0] = frag_ld(ldsA, wm + mrow, KST, koff);
    af[1] = frag_ld(ldsA, wm + 16 + mrow, KST, koff);
    b1f[0] = frag_ld(ldsB1, wn + mrow, KST, koff);
    b1f[1] = frag_ld(ldsB1, wn + 16 + mrow, KST, koff);
    b2f[0] = frag_ld(ldsB2, wn + mrow, KST, koff);
    b2f[1] = frag_ld(ldsB2, wn + 16 + mrow, KST, koff);
#pragma unroll
    for (int i = 0; i < 2; ++i)
#pragma unroll
      for (int j = 0; j < 2; ++j) {
        accg[i][j] = wmma_bf16(af[i], b1f[j], accg[i][j]);
        accu[i][j] = wmma_bf16(af[i], b2f[j], accu[i][j]);
      }
    __syncthreads();
  }

  int rbase = (lane >> 4) * 8;
  int ncol = lane & 15;
#pragma unroll
  for (int i = 0; i < 2; ++i)
#pragma unroll
    for (int j = 0; j < 2; ++j)
#pragma unroll
      for (int r = 0; r < 8; ++r) {
        long slot = mbase + wm + i * 16 + rbase + r;
        if (slot < cnt) {
          float g = accg[i][j][r];
          float u = accu[i][j][r];
          float hval = (g / (1.f + __expf(-g))) * u;
          H[((long)e * Ttok + slot) * Fm + nbase + wn + j * 16 + ncol] = (bf16)hval;
        }
      }
}

// ---------------- MoE down GEMM + weighted scatter-add ----------------
__global__ void __launch_bounds__(256)
moe_down_kernel(const bf16* __restrict__ H, const bf16* __restrict__ W2,
                const int* __restrict__ cnts, const int* __restrict__ toks,
                const float* __restrict__ wts, float* __restrict__ out) {
  int e = blockIdx.z;
  int cnt = cnts[e];
  long mbase = (long)blockIdx.y * BM;
  if (mbase >= cnt) return;
  __shared__ bf16 ldsA[BM * KST];
  __shared__ bf16 ldsB[BN * KST];
  const bf16* Ae = H + (long)e * Ttok * Fm;
  const bf16* Be = W2 + (long)e * Fm * Dm;
  int t = threadIdx.x;
  int lane = t & 31, wid = t >> 5;
  int wm = (wid & 3) * 32, wn = (wid >> 2) * 32;
  long nbase = (long)blockIdx.x * BN;
  int koff = (lane >> 4) * 8;
  int mrow = lane & 15;

  v8f acc[2][2];
#pragma unroll
  for (int i = 0; i < 2; ++i)
#pragma unroll
    for (int j = 0; j < 2; ++j) acc[i][j] = vzero8();

  for (int k0 = 0; k0 < Fm; k0 += KST) {
#pragma unroll
    for (int p = 0; p < 2; ++p) {
      int idx = t + p * 256;
      int r = idx >> 2;
      int c = (idx & 3) * 8;
#if defined(USE_ASYNC_LDS)
      async_cp16(&Ae[(mbase + r) * (long)Fm + k0 + c], &ldsA[r * KST + c]);
#else
      *(v8bf*)&ldsA[r * KST + c] = *(const v8bf*)&Ae[(mbase + r) * (long)Fm + k0 + c];
#endif
    }
    {
      int kk = t >> 3;
      int n0 = (t & 7) * 8;
      v8bf bv = *(const v8bf*)&Be[(long)(k0 + kk) * Dm + nbase + n0];
#pragma unroll
      for (int i = 0; i < 8; ++i) ldsB[(n0 + i) * KST + kk] = bv[i];
    }
#if defined(USE_ASYNC_LDS)
    async_wait0();
#endif
    __syncthreads();
    v16bf af[2], bfr[2];
    af[0] = frag_ld(ldsA, wm + mrow, KST, koff);
    af[1] = frag_ld(ldsA, wm + 16 + mrow, KST, koff);
    bfr[0] = frag_ld(ldsB, wn + mrow, KST, koff);
    bfr[1] = frag_ld(ldsB, wn + 16 + mrow, KST, koff);
#pragma unroll
    for (int i = 0; i < 2; ++i)
#pragma unroll
      for (int j = 0; j < 2; ++j) acc[i][j] = wmma_bf16(af[i], bfr[j], acc[i][j]);
    __syncthreads();
  }

  int rbase = (lane >> 4) * 8;
  int ncol = lane & 15;
#pragma unroll
  for (int i = 0; i < 2; ++i)
#pragma unroll
    for (int j = 0; j < 2; ++j)
#pragma unroll
      for (int r = 0; r < 8; ++r) {
        long slot = mbase + wm + i * 16 + rbase + r;
        if (slot < cnt) {
          int tok = toks[(long)e * Ttok + slot];
          float wt = wts[(long)e * Ttok + slot];
          long n = nbase + wn + j * 16 + ncol;
          atomicAdd(&out[(long)tok * Dm + n], wt * acc[i][j][r]);
        }
      }
}

// ==================== host-side orchestration ====================
extern "C" void kernel_launch(void* const* d_in, const int* in_sizes, int n_in,
                              void* d_out, int out_size, void* d_ws, size_t ws_size,
                              hipStream_t stream) {
  (void)in_sizes; (void)n_in; (void)out_size; (void)ws_size;
  const float* hidden   = (const float*)d_in[0];
  const float* residual = (const float*)d_in[1];
  const float* cos_t    = (const float*)d_in[2];
  const float* sin_t    = (const float*)d_in[3];
  const float* ln1_w    = (const float*)d_in[4];
  const float* ln2_w    = (const float*)d_in[5];
  const float* wq       = (const float*)d_in[6];
  const float* wk       = (const float*)d_in[7];
  const float* wv       = (const float*)d_in[8];
  const float* wo       = (const float*)d_in[9];
  const float* router_w = (const float*)d_in[10];
  const float* w1       = (const float*)d_in[11];
  const float* w3       = (const float*)d_in[12];
  const float* w2       = (const float*)d_in[13];

  char* ws = (char*)d_ws;
  size_t off = 0;
  auto take = [&](size_t bytes) {
    size_t o = off;
    off += (bytes + 255) & ~(size_t)255;
    return o;
  };

  float* res      = (float*)(ws + take((size_t)Ttok * Dm * 4));
  bf16* normed    = (bf16*)(ws + take((size_t)Ttok * Dm * 2));
  bf16* qb        = (bf16*)(ws + take((size_t)Ttok * NH * HD * 2));
  bf16* kbuf      = (bf16*)(ws + take((size_t)Ttok * NKV * HD * 2));
  bf16* vbuf      = (bf16*)(ws + take((size_t)Ttok * NKV * HD * 2));
  bf16* obuf      = (bf16*)(ws + take((size_t)Ttok * NH * HD * 2));
  float* attnproj = (float*)(ws + take((size_t)Ttok * Dm * 4));
  bf16* normed2   = (bf16*)(ws + take((size_t)Ttok * Dm * 2));
  bf16* wq_b      = (bf16*)(ws + take((size_t)Dm * NH * HD * 2));
  bf16* wk_b      = (bf16*)(ws + take((size_t)Dm * NKV * HD * 2));
  bf16* wv_b      = (bf16*)(ws + take((size_t)Dm * NKV * HD * 2));
  bf16* wo_b      = (bf16*)(ws + take((size_t)NH * HD * Dm * 2));
  bf16* w1_b      = (bf16*)(ws + take((size_t)NE * Dm * Fm * 2));
  bf16* w3_b      = (bf16*)(ws + take((size_t)NE * Dm * Fm * 2));
  bf16* w2_b      = (bf16*)(ws + take((size_t)NE * Fm * Dm * 2));
  int* cnts       = (int*)(ws + take((size_t)NE * 4));
  int* toks       = (int*)(ws + take((size_t)NE * Ttok * 4));
  float* wts      = (float*)(ws + take((size_t)NE * Ttok * 4));
  bf16* Hbuf      = (bf16*)(ws + take((size_t)NE * Ttok * Fm * 2));

  float* mlp_out  = (float*)d_out;
  float* attn_res = (float*)d_out + (size_t)Ttok * Dm;

  (void)hipMemsetAsync(mlp_out, 0, (size_t)Ttok * Dm * 4, stream);
  (void)hipMemsetAsync(cnts, 0, NE * sizeof(int), stream);

  cvt_kernel<<<4096, 256, 0, stream>>>(wq, wq_b, (long)Dm * NH * HD);
  cvt_kernel<<<2048, 256, 0, stream>>>(wk, wk_b, (long)Dm * NKV * HD);
  cvt_kernel<<<2048, 256, 0, stream>>>(wv, wv_b, (long)Dm * NKV * HD);
  cvt_kernel<<<4096, 256, 0, stream>>>(wo, wo_b, (long)NH * HD * Dm);
  cvt_kernel<<<8192, 256, 0, stream>>>(w1, w1_b, (long)NE * Dm * Fm);
  cvt_kernel<<<8192, 256, 0, stream>>>(w3, w3_b, (long)NE * Dm * Fm);
  cvt_kernel<<<8192, 256, 0, stream>>>(w2, w2_b, (long)NE * Fm * Dm);

  rmsnorm_res_kernel<<<Ttok, 256, 0, stream>>>(hidden, residual, ln1_w, res, normed);

  {
    dim3 g(NH * HD / BN, Ttok / BM);
    gemm_bf16_kernel<<<g, 256, 0, stream>>>(normed, wq_b, nullptr, qb, Ttok, NH * HD, Dm);
  }
  {
    dim3 g(NKV * HD / BN, Ttok / BM);
    gemm_bf16_kernel<<<g, 256, 0, stream>>>(normed, wk_b, nullptr, kbuf, Ttok, NKV * HD, Dm);
    gemm_bf16_kernel<<<g, 256, 0, stream>>>(normed, wv_b, nullptr, vbuf, Ttok, NKV * HD, Dm);
  }

  rope_kernel<<<((long)Ttok * NH * 64 + 255) / 256, 256, 0, stream>>>(
      qb, cos_t, sin_t, NH, (long)Ttok * NH * 64);
  rope_kernel<<<((long)Ttok * NKV * 64 + 255) / 256, 256, 0, stream>>>(
      kbuf, cos_t, sin_t, NKV, (long)Ttok * NKV * 64);

  {
    dim3 g(Bsz * NH, Sq / 64);
    flash_attn_kernel<<<g, 128, 0, stream>>>(qb, kbuf, vbuf, obuf);
  }

  {
    dim3 g(Dm / BN, Ttok / BM);
    gemm_bf16_kernel<<<g, 256, 0, stream>>>(obuf, wo_b, attnproj, nullptr, Ttok, Dm, NH * HD);
  }

  rmsnorm_res_kernel<<<Ttok, 256, 0, stream>>>(attnproj, res, ln2_w, attn_res, normed2);

  router_kernel<<<Ttok, 256, 0, stream>>>(normed2, router_w, cnts, toks, wts);

  {
    dim3 g(Fm / BN, Ttok / BM, NE);
    moe_gateup_kernel<<<g, 256, 0, stream>>>(normed2, w1_b, w3_b, cnts, toks, Hbuf);
  }
  {
    dim3 g(Dm / BN, Ttok / BM, NE);
    moe_down_kernel<<<g, 256, 0, stream>>>(Hbuf, w2_b, cnts, toks, wts, mlp_out);
  }
}
